// InvRadonLayer_35158602285223
// MI455X (gfx1250) — matrix operand order repaired and checked by reading, not compiled, not run on hardware
//
#include <hip/hip_runtime.h>
#include <hip/hip_bf16.h>
#include <stdint.h>
#include <math.h>

#ifndef M_PI
#define M_PI 3.14159265358979323846
#endif

// Problem constants: N=4, H=512, W=360, D=512
#define HN 512
#define WN 360
#define NB 4
#define DN 512
#define NJ (WN * NB)      // 1440 GEMM columns, j = w*4 + n
#define WC 15             // angles per LDS chunk; 2 buffers * 15*512*4B = 60KB
#define NCH (WN / WC)     // 24 chunks

typedef __attribute__((ext_vector_type(16))) _Float16 v16h;
typedef __attribute__((ext_vector_type(8)))  _Float16 v8h;
typedef __attribute__((ext_vector_type(8)))  float    v8f;
typedef __attribute__((ext_vector_type(4)))  float    v4f;
typedef __attribute__((ext_vector_type(4)))  unsigned int v4u;
typedef __attribute__((ext_vector_type(8)))  int      v8i;
typedef __attribute__((ext_vector_type(4)))  int      v4i;

// ---------------------------------------------------------------------------
// Prep: circulant ramp-filter matrix C[h][k] = hG[(k - h - 257) mod 512], f16
// ---------------------------------------------------------------------------
__global__ void prep_cmat(const float* __restrict__ hG, _Float16* __restrict__ Ch) {
    int e = blockIdx.x * 256 + threadIdx.x;        // 262144 elements
    int h = e >> 9, k = e & 511;
    Ch[e] = (_Float16)hG[(k - h - 257 + 1024) & 511];
}

// Prep: transposed sinogram Rt[j][k] = radon[n][k][w], j = w*4+n, f16
__global__ void prep_rt(const float* __restrict__ R, _Float16* __restrict__ Rt) {
    int e = blockIdx.x * 256 + threadIdx.x;        // 737280 elements
    int j = e >> 9, k = e & 511;
    int n = j & 3, w = j >> 2;
    Rt[e] = (_Float16)R[n * (HN * WN) + k * WN + w];
}

// Prep: per-angle (cos, sin) table, double precision to match numpy
__global__ void prep_trig(float2* __restrict__ trig) {
    int w = blockIdx.x * 256 + threadIdx.x;
    if (w < WN) {
        double th = M_PI * (double)w / 360.0;      // pi/180 * (180*w/360)
        trig[w] = make_float2((float)cos(th), (float)sin(th));
    }
}

// ---------------------------------------------------------------------------
// Filter GEMM: F[j][h] = sum_k C[h][k] * Rt[j][k]  via v_wmma_f32_16x16x32_f16
// One wave per block; each wave owns a 16x16 (h x j) output tile, K=512.
// ---------------------------------------------------------------------------
__global__ __launch_bounds__(32) void gemm_filter(const _Float16* __restrict__ Ch,
                                                  const _Float16* __restrict__ Rt,
                                                  float* __restrict__ F) {
    const int l    = threadIdx.x;
    const int j0   = blockIdx.x * 16;              // column tile (j)
    const int h0   = blockIdx.y * 16;              // row tile (h = M)
    const int lo16 = (l < 16) ? 1 : 0;
    const int row  = h0 + (l & 15);                // A-matrix row M
    const int jj   = j0 + (l & 15);                // B-matrix col N

    v8f acc = {};
    for (int k0 = 0; k0 < HN; k0 += 32) {
        // A fragment (16x32 f16): lanes<16 hold K 0..15, lanes>=16 hold K 16..31
        int kba = k0 + (lo16 ? 0 : 16);
        v8h a_lo = *(const v8h*)(Ch + row * HN + kba);
        v8h a_hi = *(const v8h*)(Ch + row * HN + kba + 8);
        v16h a;
#pragma unroll
        for (int i = 0; i < 8; ++i) { a[i] = a_lo[i]; a[i + 8] = a_hi[i]; }
        // B fragment (32x16 f16): lanes<16 hold K 0..15, lanes>=16 hold K 16..31
        int kbb = k0 + (lo16 ? 0 : 16);
        v16h b = *(const v16h*)(Rt + jj * HN + kbb);

        acc = __builtin_amdgcn_wmma_f32_16x16x32_f16(
            /*neg_a=*/false, a, /*neg_b=*/false, b,
            /*c_mod=*/(short)0, acc, /*reuse_a=*/false, /*reuse_b=*/false);
    }
    // D layout: VGPR r -> M = r + 8*(lane/16), N = lane%16. Store F[j][h].
    int mb = h0 + (lo16 ? 0 : 8);
    float* dst = F + (size_t)jj * HN + mb;
    v4f lo4 = { acc[0], acc[1], acc[2], acc[3] };
    v4f hi4 = { acc[4], acc[5], acc[6], acc[7] };
    *(v4f*)(dst)     = lo4;
    *(v4f*)(dst + 4) = hi4;
}

// ---------------------------------------------------------------------------
// TDM helper: async-load WC rows of 512 f32 (stride NB*HN elems) into LDS.
// D# per CDNA5 ISA §8: group0 = count/lds/global/type=2; group1 = dims/strides.
// ---------------------------------------------------------------------------
__device__ __forceinline__ void tdm_load_rows(const float* gptr, uint32_t lds_off) {
    uint64_t ga = (uint64_t)(uintptr_t)gptr;
    v4u g0;
    g0[0] = 1u;                                            // count=1, user mode
    g0[1] = lds_off;                                       // lds_addr (bytes)
    g0[2] = (uint32_t)(ga & 0xFFFFFFFFu);                  // global_addr lo
    g0[3] = (uint32_t)((ga >> 32) & 0x01FFFFFFu) | (2u << 30);  // addr hi | type=2
    const uint32_t td0 = HN, td1 = WC, tl0 = HN, tl1 = WC;
    const uint64_t s0  = (uint64_t)(NB * HN);              // elems between angle lines
    v8i g1;
    g1[0] = (int)(2u << 16);                               // data_size = 4B
    g1[1] = (int)((td0 & 0xFFFFu) << 16);                  // tensor_dim0 lo16
    g1[2] = (int)((td0 >> 16) | ((td1 & 0xFFFFu) << 16));  // dim0 hi | dim1 lo
    g1[3] = (int)((td1 >> 16) | (tl0 << 16));              // dim1 hi | tile_dim0
    g1[4] = (int)(tl1);                                    // tile_dim1 (tile_dim2=0)
    g1[5] = (int)(s0 & 0xFFFFFFFFu);                       // dim0_stride lo
    g1[6] = (int)((uint32_t)(s0 >> 32));                   // dim0_stride hi
    g1[7] = 0;
    v4i z4 = {0, 0, 0, 0};
#if __clang_major__ >= 23
    v8i z8 = {0, 0, 0, 0, 0, 0, 0, 0};
    __builtin_amdgcn_tensor_load_to_lds(g0, g1, z4, z4, z8, 0);
#else
    __builtin_amdgcn_tensor_load_to_lds(g0, g1, z4, z4, 0);
#endif
}

// ---------------------------------------------------------------------------
// Backprojection: per block = (4 image rows) x (one batch n).
// Double-buffered TDM: wave 0 issues chunk c+1 while block computes chunk c;
// in-order TENSORcnt completion => wait(1) guarantees chunk c is resident.
// ---------------------------------------------------------------------------
__global__ __launch_bounds__(256) void backproj(const float* __restrict__ F,
                                                const float2* __restrict__ trig,
                                                float* __restrict__ out) {
    __shared__ float smem[2][WC * HN];                     // 2 x 30KB ping-pong
    const int tid = threadIdx.x;
    const int n   = blockIdx.y;
    const int i0  = blockIdx.x * 4;                        // first image row

    float acc[8];
    float jf[8], iff[8];
#pragma unroll
    for (int k = 0; k < 8; ++k) {
        acc[k] = 0.0f;
        int p  = k * 256 + tid;                            // linear pixel in 4x512 tile
        jf[k]  = (float)((p & 511) - 256);                 // ypr = j - D/2
        iff[k] = (float)(i0 + (p >> 9) - 256);             // xpr = i - D/2
    }

    const float* Fbase = F + (size_t)n * HN;               // row j = w*NB + n
    if (tid < 32) {                                        // prologue: chunk 0
        tdm_load_rows(Fbase + (size_t)(0 * WC) * NB * HN,
                      (uint32_t)(uintptr_t)(&smem[0][0]));
    }

    for (int c = 0; c < NCH; ++c) {
        if (tid < 32) {
            if (c + 1 < NCH) {                             // issue next chunk async
                tdm_load_rows(Fbase + (size_t)((c + 1) * WC) * NB * HN,
                              (uint32_t)(uintptr_t)(&smem[(c + 1) & 1][0]));
                __builtin_amdgcn_s_wait_tensorcnt((short)1);   // chunk c resident
            } else {
                __builtin_amdgcn_s_wait_tensorcnt((short)0);   // last chunk resident
            }
        }
        __syncthreads();

        const float* buf = &smem[c & 1][0];
        const int w0 = c * WC;
        for (int wl = 0; wl < WC; ++wl) {
            const float2 cs = trig[w0 + wl];
            const float* col = buf + wl * HN;
            // py = ((j-256)*cos - (i-256)*sin) * (255.5/256) + 255.5
            const float a = cs.x * (255.5f / 256.0f);
            const float b = cs.y * (255.5f / 256.0f);
#pragma unroll
            for (int k = 0; k < 8; ++k) {
                float py = jf[k] * a - iff[k] * b + 255.5f;
                float fl = floorf(py);
                int   y0 = (int)fl;
                float fy = py - fl;
                int   y1 = y0 + 1;
                float v = 0.0f;
                if (y0 >= 0 && y0 < HN) v += col[y0] * (1.0f - fy);
                if (y1 >= 0 && y1 < HN) v += col[y1] * fy;
                acc[k] += v;
            }
        }
        __syncthreads();                                   // buffer free for reuse
    }

    const float scale = (float)(M_PI / (2.0 * WN));
    float* o = out + (size_t)n * (DN * DN) + (size_t)i0 * DN;
#pragma unroll
    for (int k = 0; k < 8; ++k) o[k * 256 + tid] = acc[k] * scale;
}

// ---------------------------------------------------------------------------
extern "C" void kernel_launch(void* const* d_in, const int* in_sizes, int n_in,
                              void* d_out, int out_size, void* d_ws, size_t ws_size,
                              hipStream_t stream) {
    const float* radon = (const float*)d_in[0];   // (4,1,512,360) f32
    const float* hG    = (const float*)d_in[1];   // (1,1,512,1)   f32
    // d_in[2] (t_y, 377MB) deliberately unused: recomputed from 360-entry trig table
    float* out = (float*)d_out;                   // (4,1,512,512) f32

    char* ws = (char*)d_ws;
    _Float16* Ch   = (_Float16*)(ws);                 // 512KB  circulant filter f16
    _Float16* Rt   = (_Float16*)(ws + 524288);        // 1.44MB transposed sinogram f16
    float*    F    = (float*)   (ws + 1998848);       // 2.95MB filtered, [j][h]
    float2*   trig = (float2*)  (ws + 4947968);       // 2.9KB  (cos,sin) per angle

    prep_cmat<<<262144 / 256, 256, 0, stream>>>(hG, Ch);
    prep_rt  <<<737280 / 256, 256, 0, stream>>>(radon, Rt);
    prep_trig<<<2, 256, 0, stream>>>(trig);

    gemm_filter<<<dim3(NJ / 16, HN / 16), 32, 0, stream>>>(Ch, Rt, F);

    backproj<<<dim3(DN / 4, NB), 256, 0, stream>>>(F, trig, out);
}
